// MoEFFN_24283745091735
// MI455X (gfx1250) — matrix operand order repaired and checked by reading, not compiled, not run on hardware
//
#include <hip/hip_runtime.h>
#include <hip/hip_bf16.h>
#include <math.h>

// ---------------- problem constants (from reference) ----------------
#define S     4096      // tokens (B*T)
#define C     1024      // embed
#define E     64        // experts
#define K_TOP 8         // router top-k
#define G     8         // groups
#define TG    4         // top-k groups
#define H     512       // expert hidden
#define HS    2048      // shared hidden

#define ROWS      (S * K_TOP)          // 32768 routed rows
#define ROWS_PAD  (ROWS + E * 128)     // per-expert 128-padding worst case = 40960
#define MAX_TILES (ROWS_PAD / 128)     // 320

// ---------------- GEMM tiling ----------------
#define BM 128
#define BN 64
#define BK 32
#define APITCH 40   // bf16 elems per LDS A row (80B -> 16B-aligned b128 reads)
#define BPITCH 40

// async global->LDS copy support (gfx1250); fall back to reg copies if absent
#if defined(__has_builtin)
#  if __has_builtin(__builtin_amdgcn_global_load_async_to_lds_b128)
#    define ASYNC_LDS 1
#  endif
#endif
#ifndef ASYNC_LDS
#  define ASYNC_LDS 0
#endif

typedef __bf16 bf16;
typedef __attribute__((ext_vector_type(2)))  bf16  bf16x2;
typedef __attribute__((ext_vector_type(4)))  bf16  bf16x4;
typedef __attribute__((ext_vector_type(8)))  bf16  bf16x8;
typedef __attribute__((ext_vector_type(16))) bf16  bf16x16;
typedef __attribute__((ext_vector_type(8)))  float f32x8;
typedef int v4i __attribute__((vector_size(4 * sizeof(int))));  // matches builtin param

static __device__ __forceinline__ bf16x16 frag16(const bf16* p0, const bf16* p1) {
  bf16x8 lo = *(const bf16x8*)p0;
  bf16x8 hi = *(const bf16x8*)p1;
  return __builtin_shufflevector(lo, hi, 0,1,2,3,4,5,6,7,8,9,10,11,12,13,14,15);
}

static __device__ __forceinline__ float silu_f(float g) {
  return g * __builtin_amdgcn_rcpf(1.f + __expf(-g));
}

// ---------------- register-staged tile loads (fp32 -> bf16 at LDS store) ----------------
struct AReg { float4 v[4]; };   // 128x32 tile, 256 threads, 4 float4 each
struct BReg { float4 a, b; };   // 32x64 tile: 2 consecutive K rows, same 4-col group

static __device__ __forceinline__ AReg ld_A(const float* A, int lda, int rowBase, int k0) {
  AReg r;
  const int tid = threadIdx.x;
#pragma unroll
  for (int j = 0; j < 4; ++j) {
    int q = tid + 256 * j;
    int rr = q >> 3, c4 = q & 7;
    r.v[j] = *(const float4*)(A + (size_t)(rowBase + rr) * lda + k0 + c4 * 4);
  }
  return r;
}

static __device__ __forceinline__ AReg ld_A_gather(const float* X, const int* row_token,
                                                   int rowBase, int k0) {
  AReg r;
  const int tid = threadIdx.x;
#pragma unroll
  for (int j = 0; j < 4; ++j) {
    int q = tid + 256 * j;
    int rr = q >> 3, c4 = q & 7;
    int tok = row_token[rowBase + rr];
    r.v[j] = *(const float4*)(X + (size_t)tok * C + k0 + c4 * 4);
  }
  return r;
}

static __device__ __forceinline__ void st_A(bf16* lA, const AReg& r) {
  const int tid = threadIdx.x;
#pragma unroll
  for (int j = 0; j < 4; ++j) {
    int q = tid + 256 * j;
    int rr = q >> 3, c4 = q & 7;
    bf16x4 b = { (bf16)r.v[j].x, (bf16)r.v[j].y, (bf16)r.v[j].z, (bf16)r.v[j].w };
    *(bf16x4*)(lA + rr * APITCH + c4 * 4) = b;
  }
}

// 256 threads = 16 k-pairs x 16 col-groups; two K rows per thread
static __device__ __forceinline__ BReg ld_B(const float* W, int ldn, int n0, int k0) {
  BReg r;
  const int tid = threadIdx.x;
  int kp = tid >> 4, c4 = tid & 15;
  const float* p = W + (size_t)(k0 + 2 * kp) * ldn + n0 + c4 * 4;
  r.a = *(const float4*)p;
  r.b = *(const float4*)(p + ldn);
  return r;
}

// packed bf16x2 (b32) stores: element pair (k, k+1) adjacent along a B^T row
static __device__ __forceinline__ void st_B(bf16* lBt, const BReg& r) {
  const int tid = threadIdx.x;
  int kp = tid >> 4, c4 = tid & 15;
  int n = c4 * 4;
  bf16* base = lBt + 2 * kp;
  bf16x2 p0 = { (bf16)r.a.x, (bf16)r.b.x };
  bf16x2 p1 = { (bf16)r.a.y, (bf16)r.b.y };
  bf16x2 p2 = { (bf16)r.a.z, (bf16)r.b.z };
  bf16x2 p3 = { (bf16)r.a.w, (bf16)r.b.w };
  *(bf16x2*)(base + (n + 0) * BPITCH) = p0;
  *(bf16x2*)(base + (n + 1) * BPITCH) = p1;
  *(bf16x2*)(base + (n + 2) * BPITCH) = p2;
  *(bf16x2*)(base + (n + 3) * BPITCH) = p3;
}

// ---------------- bf16 A tile: raw 16B copies, async-to-LDS when available ----------------
static __device__ __forceinline__ void copy_A_bf16(bf16* lA, const bf16* A, int lda,
                                                   int rowBase, int k0) {
  const int tid = threadIdx.x;
#pragma unroll
  for (int j = 0; j < 2; ++j) {
    int q = tid + 256 * j;          // 512 x 16B chunks
    int r = q >> 2, seg = q & 3;
    const bf16* g = A + (size_t)(rowBase + r) * lda + k0 + seg * 8;
    bf16* l = lA + r * APITCH + seg * 8;
#if ASYNC_LDS
    __builtin_amdgcn_global_load_async_to_lds_b128(
        (v4i*)g,                                            // C-style cast drops const
        (__attribute__((address_space(3))) v4i*)l,          // LDS destination
        0, 0);
#else
    *(bf16x8*)l = *(const bf16x8*)g;
#endif
  }
}

static __device__ __forceinline__ void wait_async() {
#if ASYNC_LDS
  asm volatile("s_wait_asynccnt 0" ::: "memory");
#endif
}

// ---------------- wave-level 32x32 WMMA step ----------------
static __device__ __forceinline__ void wave_mma(const bf16* lA, const bf16* lBt,
                                                int wm, int wn, int lane, f32x8 acc[2][2]) {
  const int l15 = lane & 15, kh = (lane >> 4) & 1;
  bf16x16 a[2], b[2];
#pragma unroll
  for (int i = 0; i < 2; ++i) {
    const bf16* p = lA + (wm * 32 + i * 16 + l15) * APITCH + kh * 8;
    a[i] = frag16(p, p + 16);
  }
#pragma unroll
  for (int j = 0; j < 2; ++j) {
    const bf16* p = lBt + (wn * 32 + j * 16 + l15) * BPITCH + kh * 8;
    b[j] = frag16(p, p + 16);
  }
#pragma unroll
  for (int i = 0; i < 2; ++i)
#pragma unroll
    for (int j = 0; j < 2; ++j)
      acc[i][j] = __builtin_amdgcn_wmma_f32_16x16x32_bf16(
          false, a[i], false, b[j], (short)0, acc[i][j], false, false);
}

// ---------------- routing ----------------
__global__ void __launch_bounds__(64)
k_router(const float* __restrict__ x, const float* __restrict__ rw,
         const float* __restrict__ ebias, int* __restrict__ topk_idx,
         float* __restrict__ topk_w) {
  __shared__ float xr[C];
  __shared__ float sc[E];
  __shared__ float sb[E];
  const int t = blockIdx.x, tid = threadIdx.x;
  const float* xrow = x + (size_t)t * C;
  for (int i = tid; i < C; i += 64) xr[i] = xrow[i];
  __syncthreads();
  const float4* w4 = (const float4*)(rw + (size_t)tid * C);
  float acc = 0.f;
#pragma unroll 4
  for (int k = 0; k < C / 4; ++k) {
    float4 t4 = w4[k];
    acc += xr[4*k] * t4.x + xr[4*k+1] * t4.y + xr[4*k+2] * t4.z + xr[4*k+3] * t4.w;
  }
  float s = __builtin_amdgcn_rcpf(1.f + __expf(-acc));
  sc[tid] = s;
  sb[tid] = s + ebias[tid];
  __syncthreads();
  if (tid == 0) {
    const float NEG = -3.4e38f;
    float grp[G];
    for (int g = 0; g < G; ++g) {              // group score = top2 biased sum
      float m1 = NEG, m2 = NEG;
      for (int i = 0; i < E / G; ++i) {
        float v = sb[g * (E / G) + i];
        if (v > m1) { m2 = m1; m1 = v; } else if (v > m2) { m2 = v; }
      }
      grp[g] = m1 + m2;
    }
    bool gsel[G]; for (int g = 0; g < G; ++g) gsel[g] = false;
    for (int r = 0; r < TG; ++r) {             // top-4 groups
      int best = 0; float bv = NEG;
      for (int g = 0; g < G; ++g) if (!gsel[g] && grp[g] > bv) { bv = grp[g]; best = g; }
      gsel[best] = true;
    }
    bool esel[E]; for (int e = 0; e < E; ++e) esel[e] = false;
    int idx[K_TOP]; float wv[K_TOP]; float wsum = 0.f;
    for (int r = 0; r < K_TOP; ++r) {          // top-8 experts within chosen groups
      int be = 0; float bv = NEG;
      for (int e = 0; e < E; ++e)
        if (gsel[e >> 3] && !esel[e] && sb[e] > bv) { bv = sb[e]; be = e; }
      esel[be] = true; idx[r] = be; wv[r] = sc[be]; wsum += sc[be];
    }
    float inv = 1.f / (wsum + 1e-20f);
    for (int r = 0; r < K_TOP; ++r) {
      topk_idx[t * K_TOP + r] = idx[r];
      topk_w[t * K_TOP + r] = wv[r] * inv;
    }
  }
}

__global__ void k_init(int* row_token, float* row_weight, int* counts, int* cursor,
                       int* tile_expert) {
  int i = blockIdx.x * blockDim.x + threadIdx.x;
  if (i < ROWS_PAD) { row_token[i] = 0; row_weight[i] = 0.f; }
  if (i < E) { counts[i] = 0; cursor[i] = 0; }
  if (i < MAX_TILES) tile_expert[i] = -1;
}

__global__ void k_count(const int* __restrict__ topk_idx, int* __restrict__ counts) {
  int i = blockIdx.x * blockDim.x + threadIdx.x;
  if (i < ROWS) atomicAdd(&counts[topk_idx[i]], 1);
}

__global__ void k_scan(const int* __restrict__ counts, int* __restrict__ aligned_off,
                       int* __restrict__ tile_expert) {
  if (threadIdx.x == 0 && blockIdx.x == 0) {
    int run = 0;
    for (int e = 0; e < E; ++e) {
      aligned_off[e] = run;
      int nt = (counts[e] + BM - 1) / BM;
      int t0 = run / BM;
      for (int j = 0; j < nt; ++j) tile_expert[t0 + j] = e;
      run += nt * BM;
    }
  }
}

__global__ void k_fill(const int* __restrict__ topk_idx, const float* __restrict__ topk_w,
                       const int* __restrict__ aligned_off, int* __restrict__ cursor,
                       int* __restrict__ row_token, float* __restrict__ row_weight,
                       int* __restrict__ posmap) {
  int i = blockIdx.x * blockDim.x + threadIdx.x;
  if (i < ROWS) {
    int e = topk_idx[i];
    int p = aligned_off[e] + atomicAdd(&cursor[e], 1);
    row_token[p] = i / K_TOP;
    row_weight[p] = topk_w[i];
    posmap[i] = p;   // deterministic combine; in-bucket placement doesn't affect numerics
  }
}

// ---------------- shared expert: gate+up fused, hs stored bf16 ----------------
__global__ void __launch_bounds__(256)
k_shared_gateup(const float* __restrict__ x, const float* __restrict__ gw,
                const float* __restrict__ uw, bf16* __restrict__ hsb) {
  __shared__ bf16 lA[2][BM * APITCH];
  __shared__ bf16 lG[2][BN * BPITCH];
  __shared__ bf16 lU[2][BN * BPITCH];
  const int rowBase = blockIdx.x * BM, colBase = blockIdx.y * BN;
  const int lane = threadIdx.x & 31, wave = threadIdx.x >> 5;
  const int wm = wave & 3, wn = wave >> 2;
  const int NK = C / BK;
  f32x8 ag[2][2] = {}, au[2][2] = {};
  AReg ra = ld_A(x, C, rowBase, 0);
  BReg rg = ld_B(gw, HS, colBase, 0);
  BReg ru = ld_B(uw, HS, colBase, 0);
  st_A(lA[0], ra); st_B(lG[0], rg); st_B(lU[0], ru);
  ra = ld_A(x, C, rowBase, BK);
  rg = ld_B(gw, HS, colBase, BK);
  ru = ld_B(uw, HS, colBase, BK);
  __syncthreads();
  for (int kt = 0; kt < NK; ++kt) {
    const int cur = kt & 1, nxt = cur ^ 1;
    if (kt + 1 < NK) {
      st_A(lA[nxt], ra); st_B(lG[nxt], rg); st_B(lU[nxt], ru);
      if (kt + 2 < NK) {
        const int k0 = (kt + 2) * BK;
        ra = ld_A(x, C, rowBase, k0);
        rg = ld_B(gw, HS, colBase, k0);
        ru = ld_B(uw, HS, colBase, k0);
      }
    }
    wave_mma(lA[cur], lG[cur], wm, wn, lane, ag);
    wave_mma(lA[cur], lU[cur], wm, wn, lane, au);
    __syncthreads();
  }
  const int l15 = lane & 15, kh = (lane >> 4) & 1;
#pragma unroll
  for (int i = 0; i < 2; ++i)
#pragma unroll
    for (int j = 0; j < 2; ++j)
#pragma unroll
      for (int v = 0; v < 8; ++v) {
        int r = rowBase + wm * 32 + i * 16 + kh * 8 + v;
        int c = colBase + wn * 32 + j * 16 + l15;
        hsb[(size_t)r * HS + c] = (bf16)(silu_f(ag[i][j][v]) * au[i][j][v]);
      }
}

__global__ void __launch_bounds__(256)
k_shared_down(const bf16* __restrict__ hsb, const float* __restrict__ dw,
              float* __restrict__ out) {
  __shared__ bf16 lA[2][BM * APITCH];
  __shared__ bf16 lB[2][BN * BPITCH];
  const int rowBase = blockIdx.x * BM, colBase = blockIdx.y * BN;
  const int lane = threadIdx.x & 31, wave = threadIdx.x >> 5;
  const int wm = wave & 3, wn = wave >> 2;
  const int NK = HS / BK;
  f32x8 acc[2][2] = {};
  copy_A_bf16(lA[0], hsb, HS, rowBase, 0);
  BReg rb = ld_B(dw, C, colBase, 0);
  st_B(lB[0], rb);
  rb = ld_B(dw, C, colBase, BK);
  wait_async();
  __syncthreads();
  for (int kt = 0; kt < NK; ++kt) {
    const int cur = kt & 1, nxt = cur ^ 1;
    if (kt + 1 < NK) {
      copy_A_bf16(lA[nxt], hsb, HS, rowBase, (kt + 1) * BK);
      st_B(lB[nxt], rb);
      if (kt + 2 < NK) rb = ld_B(dw, C, colBase, (kt + 2) * BK);
    }
    wave_mma(lA[cur], lB[cur], wm, wn, lane, acc);
    if (kt + 1 < NK) wait_async();
    __syncthreads();
  }
  const int l15 = lane & 15, kh = (lane >> 4) & 1;
#pragma unroll
  for (int i = 0; i < 2; ++i)
#pragma unroll
    for (int j = 0; j < 2; ++j)
#pragma unroll
      for (int v = 0; v < 8; ++v) {
        int r = rowBase + wm * 32 + i * 16 + kh * 8 + v;
        int c = colBase + wn * 32 + j * 16 + l15;
        out[(size_t)r * C + c] = acc[i][j][v];   // first writer of d_out
      }
}

// ---------------- routed experts ----------------
__global__ void __launch_bounds__(256)
k_moe_gateup(const float* __restrict__ x, const float* __restrict__ gate_w,
             const float* __restrict__ up_w, const int* __restrict__ tile_expert,
             const int* __restrict__ row_token, bf16* __restrict__ hb) {
  __shared__ bf16 lA[2][BM * APITCH];
  __shared__ bf16 lG[2][BN * BPITCH];
  __shared__ bf16 lU[2][BN * BPITCH];
  const int e = tile_expert[blockIdx.x];
  if (e < 0) return;                         // uniform early-exit for unused tiles
  const int rowBase = blockIdx.x * BM, colBase = blockIdx.y * BN;
  const float* gw = gate_w + (size_t)e * C * H;
  const float* uw = up_w + (size_t)e * C * H;
  const int lane = threadIdx.x & 31, wave = threadIdx.x >> 5;
  const int wm = wave & 3, wn = wave >> 2;
  const int NK = C / BK;
  f32x8 ag[2][2] = {}, au[2][2] = {};
  AReg ra = ld_A_gather(x, row_token, rowBase, 0);
  BReg rg = ld_B(gw, H, colBase, 0);
  BReg ru = ld_B(uw, H, colBase, 0);
  st_A(lA[0], ra); st_B(lG[0], rg); st_B(lU[0], ru);
  ra = ld_A_gather(x, row_token, rowBase, BK);
  rg = ld_B(gw, H, colBase, BK);
  ru = ld_B(uw, H, colBase, BK);
  __syncthreads();
  for (int kt = 0; kt < NK; ++kt) {
    const int cur = kt & 1, nxt = cur ^ 1;
    if (kt + 1 < NK) {
      st_A(lA[nxt], ra); st_B(lG[nxt], rg); st_B(lU[nxt], ru);
      if (kt + 2 < NK) {
        const int k0 = (kt + 2) * BK;
        ra = ld_A_gather(x, row_token, rowBase, k0);
        rg = ld_B(gw, H, colBase, k0);
        ru = ld_B(uw, H, colBase, k0);
      }
    }
    wave_mma(lA[cur], lG[cur], wm, wn, lane, ag);
    wave_mma(lA[cur], lU[cur], wm, wn, lane, au);
    __syncthreads();
  }
  const int l15 = lane & 15, kh = (lane >> 4) & 1;
#pragma unroll
  for (int i = 0; i < 2; ++i)
#pragma unroll
    for (int j = 0; j < 2; ++j)
#pragma unroll
      for (int v = 0; v < 8; ++v) {
        int r = rowBase + wm * 32 + i * 16 + kh * 8 + v;
        int c = colBase + wn * 32 + j * 16 + l15;
        hb[(size_t)r * H + c] = (bf16)(silu_f(ag[i][j][v]) * au[i][j][v]);
      }
}

__global__ void __launch_bounds__(256)
k_moe_down(const bf16* __restrict__ hb, const float* __restrict__ down_w,
           const int* __restrict__ tile_expert, const float* __restrict__ row_weight,
           float* __restrict__ obuf) {
  __shared__ bf16 lA[2][BM * APITCH];
  __shared__ bf16 lB[2][BN * BPITCH];
  const int e = tile_expert[blockIdx.x];
  if (e < 0) return;
  const int rowBase = blockIdx.x * BM, colBase = blockIdx.y * BN;
  const float* dw = down_w + (size_t)e * H * C;
  const int lane = threadIdx.x & 31, wave = threadIdx.x >> 5;
  const int wm = wave & 3, wn = wave >> 2;
  const int NK = H / BK;
  f32x8 acc[2][2] = {};
  copy_A_bf16(lA[0], hb, H, rowBase, 0);
  BReg rb = ld_B(dw, C, colBase, 0);
  st_B(lB[0], rb);
  rb = ld_B(dw, C, colBase, BK);
  wait_async();
  __syncthreads();
  for (int kt = 0; kt < NK; ++kt) {
    const int cur = kt & 1, nxt = cur ^ 1;
    if (kt + 1 < NK) {
      copy_A_bf16(lA[nxt], hb, H, rowBase, (kt + 1) * BK);
      st_B(lB[nxt], rb);
      if (kt + 2 < NK) rb = ld_B(dw, C, colBase, (kt + 2) * BK);
    }
    wave_mma(lA[cur], lB[cur], wm, wn, lane, acc);
    if (kt + 1 < NK) wait_async();
    __syncthreads();
  }
  const int l15 = lane & 15, kh = (lane >> 4) & 1;
#pragma unroll
  for (int i = 0; i < 2; ++i)
#pragma unroll
    for (int j = 0; j < 2; ++j)
#pragma unroll
      for (int v = 0; v < 8; ++v) {
        int r = rowBase + wm * 32 + i * 16 + kh * 8 + v;
        int c = colBase + wn * 32 + j * 16 + l15;
        obuf[(size_t)r * C + c] = acc[i][j][v] * row_weight[r];  // pad rows weight 0
      }
}

// deterministic combine: out[t] = shared[t] + sum_{k fixed order} o[pos(t,k)]
__global__ void __launch_bounds__(256)
k_combine(const float* __restrict__ obuf, const int* __restrict__ posmap,
          float* __restrict__ out) {
  __shared__ int lpos[K_TOP];
  const int t = blockIdx.x;
  if (threadIdx.x < K_TOP) lpos[threadIdx.x] = posmap[t * K_TOP + threadIdx.x];
  __syncthreads();
  const int c = threadIdx.x * 4;
  float4 v = *(float4*)(out + (size_t)t * C + c);
#pragma unroll
  for (int k = 0; k < K_TOP; ++k) {
    int p = lpos[k];
    float4 ov = *(const float4*)(obuf + (size_t)p * C + c);
    v.x += ov.x; v.y += ov.y; v.z += ov.z; v.w += ov.w;
  }
  *(float4*)(out + (size_t)t * C + c) = v;
}

extern "C" void kernel_launch(void* const* d_in, const int* in_sizes, int n_in,
                              void* d_out, int out_size, void* d_ws, size_t ws_size,
                              hipStream_t stream) {
  (void)in_sizes; (void)n_in; (void)out_size; (void)ws_size;
  const float* x        = (const float*)d_in[0];
  const float* router_w = (const float*)d_in[1];
  const float* e_bias   = (const float*)d_in[2];
  const float* gate_w   = (const float*)d_in[3];
  const float* up_w     = (const float*)d_in[4];
  const float* down_w   = (const float*)d_in[5];
  const float* sh_gate  = (const float*)d_in[6];
  const float* sh_up    = (const float*)d_in[7];
  const float* sh_down  = (const float*)d_in[8];
  float* out = (float*)d_out;

  char* ws = (char*)d_ws;
  size_t off = 0;
  auto alloc = [&](size_t bytes) -> void* {
    void* p = ws + off;
    off += (bytes + 255) & ~(size_t)255;
    return p;
  };
  int*   topk_idx    = (int*)  alloc((size_t)ROWS * 4);
  float* topk_w      = (float*)alloc((size_t)ROWS * 4);
  int*   counts      = (int*)  alloc((size_t)E * 4);
  int*   aligned_off = (int*)  alloc((size_t)E * 4);
  int*   cursor      = (int*)  alloc((size_t)E * 4);
  int*   tile_exp    = (int*)  alloc((size_t)MAX_TILES * 4);
  int*   row_token   = (int*)  alloc((size_t)ROWS_PAD * 4);
  float* row_weight  = (float*)alloc((size_t)ROWS_PAD * 4);
  int*   posmap      = (int*)  alloc((size_t)ROWS * 4);
  bf16*  hsb         = (bf16*) alloc((size_t)S * HS * 2);
  bf16*  hb          = (bf16*) alloc((size_t)ROWS_PAD * H * 2);
  float* obuf        = (float*)alloc((size_t)ROWS_PAD * C * 4);

  k_router<<<dim3(S), dim3(64), 0, stream>>>(x, router_w, e_bias, topk_idx, topk_w);
  k_init<<<dim3((ROWS_PAD + 255) / 256), dim3(256), 0, stream>>>(row_token, row_weight,
                                                                 counts, cursor, tile_exp);
  k_count<<<dim3(ROWS / 256), dim3(256), 0, stream>>>(topk_idx, counts);
  k_scan<<<dim3(1), dim3(1), 0, stream>>>(counts, aligned_off, tile_exp);
  k_fill<<<dim3(ROWS / 256), dim3(256), 0, stream>>>(topk_idx, topk_w, aligned_off, cursor,
                                                     row_token, row_weight, posmap);
  k_shared_gateup<<<dim3(S / BM, HS / BN), dim3(256), 0, stream>>>(x, sh_gate, sh_up, hsb);
  k_shared_down<<<dim3(S / BM, C / BN), dim3(256), 0, stream>>>(hsb, sh_down, out);
  k_moe_gateup<<<dim3(MAX_TILES, H / BN), dim3(256), 0, stream>>>(x, gate_w, up_w, tile_exp,
                                                                  row_token, hb);
  k_moe_down<<<dim3(MAX_TILES, C / BN), dim3(256), 0, stream>>>(hb, down_w, tile_exp,
                                                                row_weight, obuf);
  k_combine<<<dim3(S), dim3(256), 0, stream>>>(obuf, posmap, out);
}
